// AttentionBlock_64750926954927
// MI455X (gfx1250) — matrix operand order repaired and checked by reading, not compiled, not run on hardware
//
#include <hip/hip_runtime.h>
#include <hip/hip_bf16.h>
#include <stdint.h>

#define BB 8
#define CC 512
#define NN 4096
#define GG 32

typedef __attribute__((ext_vector_type(16))) __bf16        v16bf;
typedef __attribute__((ext_vector_type(8)))  float         v8f;
typedef __attribute__((ext_vector_type(4)))  unsigned int  v4u;
typedef int v4i __attribute__((vector_size(16)));   // matches async-LDS builtin param

union BFV { v16bf v; v4u u[2]; };

// CDNA5 async global->LDS copy path (guarded: falls back to vector copy)
#if defined(__has_builtin)
#  if __has_builtin(__builtin_amdgcn_global_load_async_to_lds_b128)
#    define HAVE_ASYNC_LDS 1
#  endif
#endif

// float -> bf16 round-to-nearest-even
__device__ __forceinline__ uint16_t f2bf(float f) {
    unsigned int u = __float_as_uint(f);
    u += 0x7fffu + ((u >> 16) & 1u);
    return (uint16_t)(u >> 16);
}

// B-tile fragment: this lane's 16 contiguous K values (32 bytes)
__device__ __forceinline__ v16bf load_bfB(const uint16_t* p) {
    BFV t; t.u[0] = *(const v4u*)p; t.u[1] = *(const v4u*)(p + 8); return t.v;
}
// A-tile fragment: this lane's 16 contiguous K values (32 bytes)
// (caller adds +16 elements for lanes 16..31)
__device__ __forceinline__ v16bf load_bfA(const uint16_t* p) {
    BFV t; t.u[0] = *(const v4u*)p; t.u[1] = *(const v4u*)(p + 8); return t.v;
}

__device__ __forceinline__ v8f wmma_bf16(v16bf a, v16bf b, v8f c) {
    return __builtin_amdgcn_wmma_f32_16x16x32_bf16(false, a, false, b,
                                                   (short)0, c, false, false);
}

// ---------------------------------------------------------------------------
// GroupNorm: x[B,C,N] f32 -> hT[B,N,C] bf16 (transposed for GEMM A/B tiles)
// ---------------------------------------------------------------------------
__global__ void gn_kernel(const float* __restrict__ x, const float* __restrict__ gamma,
                          const float* __restrict__ beta, uint16_t* __restrict__ hT) {
    const int g = blockIdx.x, b = blockIdx.y, tid = threadIdx.x;
    const int CPG = CC / GG;            // 16 channels per group
    const int GE  = CPG * NN;           // 65536 elements per group
    const float* xg = x + ((size_t)b * CC + (size_t)g * CPG) * NN;

    float s1 = 0.f, s2 = 0.f;
    for (int i = tid; i < GE; i += 256) { float v = xg[i]; s1 += v; s2 += v * v; }
    __shared__ float r1[256], r2[256];
    r1[tid] = s1; r2[tid] = s2; __syncthreads();
    for (int off = 128; off > 0; off >>= 1) {
        if (tid < off) { r1[tid] += r1[tid + off]; r2[tid] += r2[tid + off]; }
        __syncthreads();
    }
    const float mu  = r1[0] / (float)GE;
    const float var = r2[0] / (float)GE - mu * mu;
    const float rs  = rsqrtf(var + 1e-5f);

    for (int i = tid; i < GE; i += 256) {
        int cl = i & (CPG - 1);          // channel-in-group (fast among threads)
        int n  = i >> 4;
        int c  = g * CPG + cl;
        float v = xg[(size_t)cl * NN + n];
        float h = (v - mu) * rs * gamma[c] + beta[c];
        hT[((size_t)b * NN + n) * CC + c] = f2bf(h);
    }
}

// ---------------------------------------------------------------------------
// Weights f32 -> bf16 (wq|wk|wv|wp concatenated)
// ---------------------------------------------------------------------------
__global__ void wconv_kernel(const float* __restrict__ wq, const float* __restrict__ wk,
                             const float* __restrict__ wv, const float* __restrict__ wp,
                             uint16_t* __restrict__ dst) {
    int i = blockIdx.x * 256 + threadIdx.x;        // 0 .. 4*512*512
    const float* srcs[4] = { wq, wk, wv, wp };
    int m = i >> 18;                               // 512*512 = 2^18
    int j = i & ((CC * CC) - 1);
    dst[i] = f2bf(srcs[m][j]);
}

// ---------------------------------------------------------------------------
// Q,K projection: q[n,o] = sum_c hT[n,c] * W[o,c] + bias  -> [B,N,C] bf16
// 4 waves per WG, each wave one 16x16 (n,o) tile, A shared between Q and K.
// ---------------------------------------------------------------------------
__global__ void __launch_bounds__(128)
qkproj_kernel(const uint16_t* __restrict__ hT, const uint16_t* __restrict__ wqb,
              const uint16_t* __restrict__ wkb, const float* __restrict__ bq,
              const float* __restrict__ bk, uint16_t* __restrict__ q,
              uint16_t* __restrict__ k) {
    const int b    = blockIdx.z;
    const int n0   = blockIdx.y * 32 + (threadIdx.x >> 6) * 16;
    const int o0   = blockIdx.x * 32 + ((threadIdx.x >> 5) & 1) * 16;
    const int lane = threadIdx.x & 31;
    const int col  = lane & 15, half = lane >> 4;

    const uint16_t* arow  = hT  + ((size_t)b * NN + n0 + col) * CC + half * 16;
    const uint16_t* bqrow = wqb + (size_t)(o0 + col) * CC + half * 16;
    const uint16_t* bkrow = wkb + (size_t)(o0 + col) * CC + half * 16;

    v8f cq = {}, ck = {};
    #pragma unroll
    for (int t = 0; t < 16; ++t) {                 // K = 512 in chunks of 32
        v16bf a = load_bfA(arow + t * 32);
        cq = wmma_bf16(a, load_bfB(bqrow + t * 32), cq);
        ck = wmma_bf16(a, load_bfB(bkrow + t * 32), ck);
    }
    const float biasq = bq[o0 + col], biask = bk[o0 + col];
    #pragma unroll
    for (int r = 0; r < 8; ++r) {                  // C-tile: M = r+8*half, N = col
        int n = n0 + r + 8 * half;
        size_t idx = ((size_t)b * NN + n) * CC + o0 + col;
        q[idx] = f2bf(cq[r] + biasq);
        k[idx] = f2bf(ck[r] + biask);
    }
}

// ---------------------------------------------------------------------------
// V projection stored transposed: vT[o,n] = sum_c W[o,c]*hT[n,c] + bv -> [B,C,N]
// ---------------------------------------------------------------------------
__global__ void __launch_bounds__(128)
vproj_kernel(const uint16_t* __restrict__ hT, const uint16_t* __restrict__ wvb,
             const float* __restrict__ bv, uint16_t* __restrict__ vT) {
    const int b    = blockIdx.z;
    const int o0   = blockIdx.x * 32 + (threadIdx.x >> 6) * 16;
    const int n0   = blockIdx.y * 32 + ((threadIdx.x >> 5) & 1) * 16;
    const int lane = threadIdx.x & 31;
    const int col  = lane & 15, half = lane >> 4;

    const uint16_t* arow = wvb + (size_t)(o0 + col) * CC + half * 16;
    const uint16_t* brow = hT  + ((size_t)b * NN + n0 + col) * CC + half * 16;

    v8f acc = {};
    #pragma unroll
    for (int t = 0; t < 16; ++t)
        acc = wmma_bf16(load_bfA(arow + t * 32), load_bfB(brow + t * 32), acc);

    #pragma unroll
    for (int r = 0; r < 8; ++r) {
        int o = o0 + r + 8 * half;
        vT[((size_t)b * CC + o) * NN + n0 + col] = f2bf(acc[r] + bv[o]);
    }
}

// ---------------------------------------------------------------------------
// Flash attention: 1 WG (4 waves) per (batch, 16-query tile).
// Wave w: S columns [w*32, w*32+32) of each 128-key block, PV channels
// [w*128, (w+1)*128). Online softmax; P staged via LDS (C-layout -> A-layout).
// ---------------------------------------------------------------------------
__global__ void __launch_bounds__(128)
flash_kernel(const uint16_t* __restrict__ q, const uint16_t* __restrict__ k,
             const uint16_t* __restrict__ vT, uint16_t* __restrict__ attn) {
    const int b    = blockIdx.y;
    const int n0   = blockIdx.x * 16;
    const int tid  = threadIdx.x;
    const int w    = tid >> 5, lane = tid & 31;
    const int col  = lane & 15, half = lane >> 4;

    __shared__ __align__(16) uint16_t Qs[16 * CC];   // 16 KB query tile
    __shared__ __align__(16) uint16_t Ps[16 * 128];  // 4 KB  P block
    __shared__ float pmax[4][16], psum[4][16];
    __shared__ float sM[16], sMnew[16], sFac[16], sl[16];

    {   // Q tile staging: rows n0..n0+15 are one contiguous 16 KB span in [B,N,C]
#ifdef HAVE_ASYNC_LDS
        const char* gsrc = (const char*)(q + ((size_t)b * NN + n0) * CC);
        char* lds = (char*)Qs;
        for (int i = tid; i < (16 * CC * 2) / 16; i += 128)
            __builtin_amdgcn_global_load_async_to_lds_b128(
                (__attribute__((address_space(1))) v4i*)(gsrc + (size_t)i * 16),
                (__attribute__((address_space(3))) v4i*)(lds + (size_t)i * 16),
                0, 0);
#  if __has_builtin(__builtin_amdgcn_s_wait_asynccnt)
        __builtin_amdgcn_s_wait_asynccnt(0);
#  else
        asm volatile("s_wait_asynccnt 0x0" ::: "memory");
#  endif
#else
        const v4u* src = (const v4u*)(q + ((size_t)b * NN + n0) * CC);
        v4u* dst = (v4u*)Qs;
        for (int i = tid; i < (16 * CC) / 8; i += 128) dst[i] = src[i];
#endif
    }
    if (tid < 16) { sM[tid] = -__builtin_inff(); sl[tid] = 0.f; }
    __syncthreads();

    v8f acc[8] = {};                                 // 16 x 128 f32 accumulator
    const float SCL = 0.0637586265f;                 // log2(e) / sqrt(512)

    for (int kb = 0; kb < NN; kb += 128) {
        // ---- S = Q K^T for this wave's 32 key columns (full C=512)
        v8f s0 = {}, s1 = {};
        const uint16_t* k0row = k + ((size_t)b * NN + kb + w * 32 + col) * CC + half * 16;
        const uint16_t* k1row = k0row + (size_t)16 * CC;
        const uint16_t* aQ    = Qs + (size_t)col * CC + half * 16;
        #pragma unroll
        for (int t = 0; t < 16; ++t) {
            v16bf a = load_bfA(aQ + t * 32);
            s0 = wmma_bf16(a, load_bfB(k0row + t * 32), s0);
            s1 = wmma_bf16(a, load_bfB(k1row + t * 32), s1);
        }
        if (kb + 128 < NN) __builtin_prefetch(k0row + (size_t)128 * CC, 0, 1);

        // ---- per-row max over this wave's 32 columns (16-lane xor reduce)
        float mx[8];
        #pragma unroll
        for (int r = 0; r < 8; ++r) {
            float v = fmaxf(s0[r], s1[r]);
            #pragma unroll
            for (int m = 1; m < 16; m <<= 1) v = fmaxf(v, __shfl_xor(v, m, 32));
            mx[r] = v;
        }
        if (col == 0) {
            #pragma unroll
            for (int r = 0; r < 8; ++r) pmax[w][half * 8 + r] = mx[r];
        }
        __syncthreads();

        if (tid < 16) {
            float mn = sM[tid];
            #pragma unroll
            for (int ww = 0; ww < 4; ++ww) mn = fmaxf(mn, pmax[ww][tid]);
            sMnew[tid] = mn;
            sFac[tid]  = exp2f((sM[tid] - mn) * SCL);  // 0 on first block
        }
        __syncthreads();

        // ---- P = exp2((S - m)*SCL), rescale acc, partial row sums
        float fr[8], mr[8];
        #pragma unroll
        for (int r = 0; r < 8; ++r) { fr[r] = sFac[half * 8 + r]; mr[r] = sMnew[half * 8 + r]; }
        #pragma unroll
        for (int ct = 0; ct < 8; ++ct)
            #pragma unroll
            for (int r = 0; r < 8; ++r) acc[ct][r] *= fr[r];

        #pragma unroll
        for (int r = 0; r < 8; ++r) {
            float p0 = exp2f((s0[r] - mr[r]) * SCL);
            float p1 = exp2f((s1[r] - mr[r]) * SCL);
            int row = half * 8 + r;
            Ps[row * 128 + w * 32 + col]      = f2bf(p0);
            Ps[row * 128 + w * 32 + 16 + col] = f2bf(p1);
            float v = p0 + p1;
            #pragma unroll
            for (int m = 1; m < 16; m <<= 1) v += __shfl_xor(v, m, 32);
            mx[r] = v;
        }
        if (col == 0) {
            #pragma unroll
            for (int r = 0; r < 8; ++r) psum[w][half * 8 + r] = mx[r];
        }
        __syncthreads();

        if (tid < 16) {
            float l = sl[tid] * sFac[tid];
            #pragma unroll
            for (int ww = 0; ww < 4; ++ww) l += psum[ww][tid];
            sl[tid] = l; sM[tid] = sMnew[tid];
        }

        // ---- PV: acc(16 x 128) += P(16 x 128) * V^T-slice
        #pragma unroll
        for (int kc = 0; kc < 4; ++kc) {
            v16bf a = load_bfA(Ps + col * 128 + kc * 32 + half * 16);
            #pragma unroll
            for (int ct = 0; ct < 8; ++ct) {
                const uint16_t* vrow =
                    vT + ((size_t)b * CC + w * 128 + ct * 16 + col) * NN
                       + kb + kc * 32 + half * 16;
                acc[ct] = wmma_bf16(a, load_bfB(vrow), acc[ct]);
            }
        }
        __syncthreads();
    }

    // ---- normalize and store attn_out [B,N,C] bf16
    float inv[8];
    #pragma unroll
    for (int r = 0; r < 8; ++r) inv[r] = 1.0f / sl[half * 8 + r];
    #pragma unroll
    for (int ct = 0; ct < 8; ++ct)
        #pragma unroll
        for (int r = 0; r < 8; ++r) {
            int row = half * 8 + r;
            attn[((size_t)b * NN + n0 + row) * CC + w * 128 + ct * 16 + col] =
                f2bf(acc[ct][r] * inv[r]);
        }
}

// ---------------------------------------------------------------------------
// Output projection + bias + residual: out[b,o,n] = x + Wp*attn + bp  (f32)
// ---------------------------------------------------------------------------
__global__ void __launch_bounds__(128)
projres_kernel(const uint16_t* __restrict__ attn, const uint16_t* __restrict__ wpb,
               const float* __restrict__ bp, const float* __restrict__ x,
               float* __restrict__ out) {
    const int b    = blockIdx.z;
    const int o0   = blockIdx.x * 32 + (threadIdx.x >> 6) * 16;
    const int n0   = blockIdx.y * 32 + ((threadIdx.x >> 5) & 1) * 16;
    const int lane = threadIdx.x & 31;
    const int col  = lane & 15, half = lane >> 4;

    const uint16_t* arow = wpb  + (size_t)(o0 + col) * CC + half * 16;
    const uint16_t* brow = attn + ((size_t)b * NN + n0 + col) * CC + half * 16;

    v8f acc = {};
    #pragma unroll
    for (int t = 0; t < 16; ++t)
        acc = wmma_bf16(load_bfA(arow + t * 32), load_bfB(brow + t * 32), acc);

    #pragma unroll
    for (int r = 0; r < 8; ++r) {
        int o = o0 + r + 8 * half;
        size_t idx = ((size_t)b * CC + o) * NN + n0 + col;
        out[idx] = x[idx] + acc[r] + bp[o];
    }
}

// ---------------------------------------------------------------------------
extern "C" void kernel_launch(void* const* d_in, const int* in_sizes, int n_in,
                              void* d_out, int out_size, void* d_ws, size_t ws_size,
                              hipStream_t stream) {
    (void)in_sizes; (void)n_in; (void)out_size; (void)ws_size;
    const float* x     = (const float*)d_in[0];
    const float* gamma = (const float*)d_in[1];
    const float* beta  = (const float*)d_in[2];
    const float* wq = (const float*)d_in[3]; const float* bq = (const float*)d_in[4];
    const float* wk = (const float*)d_in[5]; const float* bk = (const float*)d_in[6];
    const float* wv = (const float*)d_in[7]; const float* bv = (const float*)d_in[8];
    const float* wp = (const float*)d_in[9]; const float* bp = (const float*)d_in[10];
    float* out = (float*)d_out;

    // Workspace layout (bf16): hT | q | k | vT | attn (32 MB each) | weights (2 MB)
    char* ws = (char*)d_ws;
    const size_t SZ = (size_t)BB * NN * CC * sizeof(uint16_t);
    uint16_t* hT   = (uint16_t*)(ws);
    uint16_t* qB   = (uint16_t*)(ws + 1 * SZ);
    uint16_t* kB   = (uint16_t*)(ws + 2 * SZ);
    uint16_t* vT   = (uint16_t*)(ws + 3 * SZ);
    uint16_t* attn = (uint16_t*)(ws + 4 * SZ);
    uint16_t* wb   = (uint16_t*)(ws + 5 * SZ);
    uint16_t* wqb = wb;
    uint16_t* wkb = wb + (size_t)CC * CC;
    uint16_t* wvb = wb + (size_t)2 * CC * CC;
    uint16_t* wpb = wb + (size_t)3 * CC * CC;

    wconv_kernel<<<4096, 256, 0, stream>>>(wq, wk, wv, wp, wb);
    gn_kernel<<<dim3(GG, BB), 256, 0, stream>>>(x, gamma, beta, hT);
    qkproj_kernel<<<dim3(CC / 32, NN / 32, BB), 128, 0, stream>>>(hT, wqb, wkb, bq, bk, qB, kB);
    vproj_kernel<<<dim3(CC / 32, NN / 32, BB), 128, 0, stream>>>(hT, wvb, bv, vT);
    flash_kernel<<<dim3(NN / 16, BB), 128, 0, stream>>>(qB, kB, vT, attn);
    projres_kernel<<<dim3(CC / 32, NN / 32, BB), 128, 0, stream>>>(attn, wpb, bp, x, out);
}
